// Block_36464272343212
// MI455X (gfx1250) — matrix operand order repaired and checked by reading, not compiled, not run on hardware
//
#include <hip/hip_runtime.h>

#define TOKENS 4096   // B*S
#define DMODEL 512
#define SEQ    2048
#define NHEAD  8
#define HDIM   64
#define DFF    2048

typedef __attribute__((ext_vector_type(16))) __bf16 v16bf;
typedef __attribute__((ext_vector_type(8)))  float  v8f;

// ---------------------------------------------------------------- helpers

__device__ __forceinline__ __bf16 f2bf(float f) {
  unsigned u = __builtin_bit_cast(unsigned, f);
  unsigned r = u + 0x7FFFu + ((u >> 16) & 1u);   // round-to-nearest-even
  unsigned short h = (unsigned short)(r >> 16);
  return __builtin_bit_cast(__bf16, h);
}

__device__ __forceinline__ v8f wmma_bf16(v16bf a, v16bf b, v8f c) {
  return __builtin_amdgcn_wmma_f32_16x16x32_bf16(
      false, a, false, b, (short)0, c, false, false);
}

// A fragment (16x32) from row-major source; vectorizes to 2x global_load_b128.
// lane r(0..15)/r+16 holds row r; element i -> k = half*16 + i.
__device__ __forceinline__ v16bf load_a_frag(const __bf16* __restrict__ A,
                                             int lda, int lane) {
  int r = lane & 15, h = (lane >> 4) & 1;
  const __bf16* base = A + (size_t)r * lda;
  v16bf a;
#pragma unroll
  for (int v = 0; v < 8; ++v) {
    int k = (h << 4) + (v << 1);
    a[2 * v]     = base[k];
    a[2 * v + 1] = base[k + 1];
  }
  return a;
}

// Swizzled fragment load: each lane reads 16 contiguous bf16 (32B, coalesced).
__device__ __forceinline__ v16bf load_frag_swz(const __bf16* __restrict__ base,
                                               size_t frag_idx, int lane) {
  return *(const v16bf*)(base + (frag_idx * 32 + lane) * 16);
}

// Swizzled B index for logical element (k,n) of a KxN B-matrix, NT16 = N/16:
//   kt=k>>5, lane=((k>>4)&1)*16 + (n&15), i=k&15, nt=n>>4
//   idx = ((kt*NT16+nt)*32 + lane)*16 + i
// Swizzled A index for logical element (r,k) of a 16x32 A-tile:
//   lane=((k>>4)&1)*16 + r, i=k&15

// 32x64 output tile per wave: 2 A-frags, 4 B-frags, 8 WMMAs per K-step.
__device__ __forceinline__ void gemm_tile_32x64(const __bf16* __restrict__ A, int lda,
                                                const __bf16* __restrict__ Bs,
                                                int NT16, int nt0,
                                                int K, int lane, v8f acc[8]) {
  for (int k0 = 0, kt = 0; k0 < K; k0 += 32, ++kt) {
    v16bf a0 = load_a_frag(A + k0, lda, lane);
    v16bf a1 = load_a_frag(A + (size_t)16 * lda + k0, lda, lane);
#pragma unroll
    for (int j = 0; j < 4; ++j) {
      v16bf b = load_frag_swz(Bs, (size_t)kt * NT16 + nt0 + j, lane);
      acc[j]     = wmma_bf16(a0, b, acc[j]);
      acc[4 + j] = wmma_bf16(a1, b, acc[4 + j]);
    }
  }
}

// ---------------------------------------------------------------- converts

// Generic fp32 [K][N] -> bf16 B-fragment-swizzled.
__global__ void __launch_bounds__(256) cvt_swz_kernel(const float* __restrict__ W,
                                                      __bf16* __restrict__ out,
                                                      int K, int N) {
  int NT16 = N >> 4;
  int total = K * N;
  int stride = gridDim.x * blockDim.x;
  for (int d = blockIdx.x * blockDim.x + threadIdx.x; d < total; d += stride) {
    int i = d & 15;
    int lane = (d >> 4) & 31;
    int tile = d >> 9;
    int nt = tile % NT16, kt = tile / NT16;
    int k = kt * 32 + ((lane >> 4) << 4) + i;
    int col = nt * 16 + (lane & 15);
    out[d] = f2bf(W[(size_t)k * N + col]);
  }
}

// Wq/Wk/Wv [H,D,HD] -> fused [D,1536] B-swizzled bf16.
__global__ void __launch_bounds__(256) cvt_qkv_kernel(const float* __restrict__ Wq,
                                                      const float* __restrict__ Wk,
                                                      const float* __restrict__ Wv,
                                                      __bf16* __restrict__ out) {
  const int NT16 = (3 * DMODEL) >> 4;   // 96
  int total = DMODEL * 3 * DMODEL;
  int stride = gridDim.x * blockDim.x;
  for (int d = blockIdx.x * blockDim.x + threadIdx.x; d < total; d += stride) {
    int i = d & 15;
    int lane = (d >> 4) & 31;
    int tile = d >> 9;
    int nt = tile % NT16, kt = tile / NT16;
    int k = kt * 32 + ((lane >> 4) << 4) + i;        // model dim
    int col = nt * 16 + (lane & 15);                 // fused qkv column
    int sec = col >> 9;
    int hh  = (col & 511) >> 6;
    int e   = col & 63;
    const float* W = (sec == 0) ? Wq : (sec == 1) ? Wk : Wv;
    out[d] = f2bf(W[((size_t)hh * DMODEL + k) * HDIM + e]);
  }
}

// ---------------------------------------------------------------- layernorm

__global__ void __launch_bounds__(256) ln_kernel(const float* __restrict__ x,
                                                 const float* __restrict__ g,
                                                 const float* __restrict__ be,
                                                 __bf16* __restrict__ out) {
  __shared__ float red[256];
  int row = blockIdx.x, t = threadIdx.x;
  const float* xr = x + (size_t)row * DMODEL;
  float v0 = xr[t], v1 = xr[t + 256];
  red[t] = v0 + v1;
  __syncthreads();
  for (int off = 128; off > 0; off >>= 1) {
    if (t < off) red[t] += red[t + off];
    __syncthreads();
  }
  float mean = red[0] * (1.0f / DMODEL);
  __syncthreads();
  float d0 = v0 - mean, d1 = v1 - mean;
  red[t] = d0 * d0 + d1 * d1;
  __syncthreads();
  for (int off = 128; off > 0; off >>= 1) {
    if (t < off) red[t] += red[t + off];
    __syncthreads();
  }
  float rstd = rsqrtf(red[0] * (1.0f / DMODEL) + 1e-5f);
  __bf16* orow = out + (size_t)row * DMODEL;
  orow[t]       = f2bf(d0 * rstd * g[t] + be[t]);
  orow[t + 256] = f2bf(d1 * rstd * g[t + 256] + be[t + 256]);
}

// ---------------------------------------------------------------- QKV GEMM
// Produces q in A-frag-swizzled, kT/v in B-frag-swizzled per-(b,h) layouts.

__global__ void __launch_bounds__(256) gemm_qkv_kernel(
    const __bf16* __restrict__ xn, const __bf16* __restrict__ W,
    const float* __restrict__ bq, const float* __restrict__ bk,
    const float* __restrict__ bv,
    __bf16* __restrict__ q, __bf16* __restrict__ kT, __bf16* __restrict__ v) {
  const int NT = (3 * DMODEL) / 64;  // 24 column tiles of 64
  int gw = (blockIdx.x * blockDim.x + threadIdx.x) >> 5;
  int lane = threadIdx.x & 31;
  int mt = gw / NT, nt = gw - mt * NT;
  if (mt >= TOKENS / 32) return;   // uniform per wave

  v8f z = {};
  v8f acc[8] = {z, z, z, z, z, z, z, z};
  gemm_tile_32x64(xn + (size_t)(mt * 32) * DMODEL, DMODEL,
                  W, 3 * DMODEL / 16, nt * 4, DMODEL, lane, acc);

  int n0 = nt * 64;
  int sec  = n0 >> 9;
  int head = (n0 & 511) >> 6;
  const float* bias = (sec == 0) ? bq : (sec == 1) ? bk : bv;
  int n = lane & 15, h = lane >> 4;
#pragma unroll
  for (int rt = 0; rt < 2; ++rt) {
#pragma unroll
    for (int vv = 0; vv < 8; ++vv) {
      int m  = mt * 32 + rt * 16 + vv + 8 * h;
      int bb = m >> 11;          // / SEQ
      int s  = m & (SEQ - 1);
      size_t bh_ = (size_t)(bb * NHEAD + head);
#pragma unroll
      for (int j = 0; j < 4; ++j) {
        int e = j * 16 + n;
        __bf16 val = f2bf(acc[rt * 4 + j][vv] + bias[head * HDIM + e]);
        if (sec == 0) {
          // q: A-frag swizzle, tile = s>>4, frag = e>>5
          int la = (((e >> 4) & 1) << 4) + (s & 15);
          int ii = e & 15;
          q[bh_ * SEQ * HDIM +
            (((size_t)(s >> 4) * 2 + (e >> 5)) * 32 + la) * 16 + ii] = val;
        } else if (sec == 1) {
          // kT: B-frag swizzle of logical B[k=e][n=s], NT16 = SEQ/16 = 128
          int lb = (((e >> 4) & 1) << 4) + (s & 15);
          kT[bh_ * HDIM * SEQ +
             (((size_t)(e >> 5) * (SEQ / 16) + (s >> 4)) * 32 + lb) * 16 +
             (e & 15)] = val;
        } else {
          // v: B-frag swizzle of logical B[k=s][n=e], NT16 = HDIM/16 = 4
          int lb = (((s >> 4) & 1) << 4) + (e & 15);
          v[bh_ * SEQ * HDIM +
            (((size_t)(s >> 5) * (HDIM / 16) + (e >> 4)) * 32 + lb) * 16 +
            (s & 15)] = val;
        }
      }
    }
  }
}

// ---------------------------------------------------------------- flash attention
// One wave = one 16-query tile of one (b,h). Online softmax, 32 keys / iter.
// All operands pre-swizzled: every fragment is one contiguous 32B load/lane.

__global__ void __launch_bounds__(128) attn_kernel(
    const __bf16* __restrict__ qg, const __bf16* __restrict__ kTg,
    const __bf16* __restrict__ vg, __bf16* __restrict__ og) {
  __shared__ __attribute__((aligned(32))) __bf16 pstage[4][16 * 32];
  int wave = threadIdx.x >> 5;
  int lane = threadIdx.x & 31;
  int gw = blockIdx.x * 4 + wave;
  int qt = gw & (SEQ / 16 - 1);  // 128 q-tiles per (b,h)
  int bh = gw >> 7;              // 0..15
  const __bf16* qs  = qg  + (size_t)bh * SEQ * HDIM;
  const __bf16* kTs = kTg + (size_t)bh * HDIM * SEQ;
  const __bf16* vs  = vg  + (size_t)bh * SEQ * HDIM;

  // q A-frags: tile qt, frags 0/1 (K=0..31 / 32..63)
  v16bf a0 = load_frag_swz(qs, (size_t)qt * 2 + 0, lane);
  v16bf a1 = load_frag_swz(qs, (size_t)qt * 2 + 1, lane);

  v8f z = {};
  v8f o0 = z, o1 = z, o2 = z, o3 = z;
  float mrow[8], lrow[8];
#pragma unroll
  for (int i = 0; i < 8; ++i) { mrow[i] = -3.0e38f; lrow[i] = 0.0f; }

  __bf16* pt = &pstage[wave][0];
  int n = lane & 15, h = lane >> 4;
  int pl = ((n >> 3) << 4);   // table-order lane group; unused
  int pi = n & 7;

  for (int t0 = 0; t0 < SEQ; t0 += 32) {
    int ntk = t0 >> 4;
    // scores: S[16 x 32] = q(16x64) . kT(64keys-dim x keys)
    v8f s0 = z, s1 = z;
    s0 = wmma_bf16(a0, load_frag_swz(kTs, (size_t)0 * 128 + ntk,     lane), s0);
    s0 = wmma_bf16(a1, load_frag_swz(kTs, (size_t)1 * 128 + ntk,     lane), s0);
    s1 = wmma_bf16(a0, load_frag_swz(kTs, (size_t)0 * 128 + ntk + 1, lane), s1);
    s1 = wmma_bf16(a1, load_frag_swz(kTs, (size_t)1 * 128 + ntk + 1, lane), s1);

    // online softmax: each row lives on 16 lanes of one half -> masks 1,2,4,8
#pragma unroll
    for (int vv = 0; vv < 8; ++vv) {
      float sm = fmaxf(s0[vv], s1[vv]);
#pragma unroll
      for (int msk = 1; msk < 16; msk <<= 1)
        sm = fmaxf(sm, __shfl_xor(sm, msk, 32));
      float mnew = fmaxf(mrow[vv], sm);
      float corr = __expf(mrow[vv] - mnew);
      float p0 = __expf(s0[vv] - mnew);
      float p1 = __expf(s1[vv] - mnew);
      float ps = p0 + p1;
#pragma unroll
      for (int msk = 1; msk < 16; msk <<= 1)
        ps += __shfl_xor(ps, msk, 32);
      lrow[vv] = lrow[vv] * corr + ps;
      mrow[vv] = mnew;
      o0[vv] *= corr; o1[vv] *= corr; o2[vv] *= corr; o3[vv] *= corr;
      s0[vv] = p0; s1[vv] = p1;
    }

    // stage P into LDS directly in A-frag order -> consumer does 1 vector load
#pragma unroll
    for (int vv = 0; vv < 8; ++vv) {
      int row = vv + 8 * h;
      pt[row * 16 + n]        = f2bf(s0[vv]);  // cols 0..15
      pt[(16 + row) * 16 + n] = f2bf(s1[vv]);  // cols 16..31
    }
    asm volatile("s_wait_dscnt 0x0\n\ts_wait_storecnt 0x0" ::: "memory");
    v16bf pa = *(const v16bf*)(pt + lane * 16);

    // O += P(16x32) . V(32keys x 64)
    size_t ktv = (size_t)(t0 >> 5) * 4;
    o0 = wmma_bf16(pa, load_frag_swz(vs, ktv + 0, lane), o0);
    o1 = wmma_bf16(pa, load_frag_swz(vs, ktv + 1, lane), o1);
    o2 = wmma_bf16(pa, load_frag_swz(vs, ktv + 2, lane), o2);
    o3 = wmma_bf16(pa, load_frag_swz(vs, ktv + 3, lane), o3);
  }

  // finalize: /l, write concat-head row-major bf16 [token][h*64+e]
  int b_ = bh >> 3, head = bh & 7;
#pragma unroll
  for (int vv = 0; vv < 8; ++vv) {
    int s = qt * 16 + vv + 8 * h;
    float inv = 1.0f / lrow[vv];
    __bf16* dst = og + ((size_t)(b_ * SEQ + s)) * DMODEL + head * HDIM + n;
    dst[0]  = f2bf(o0[vv] * inv);
    dst[16] = f2bf(o1[vv] * inv);
    dst[32] = f2bf(o2[vv] * inv);
    dst[48] = f2bf(o3[vv] * inv);
  }
}

// ---------------------------------------------------------------- generic GEMMs

// out[f32, M x N] = res + A.Bswz + bias   (proj + MLP2)
__global__ void __launch_bounds__(256) gemm_bias_res_kernel(
    const __bf16* __restrict__ A, int lda,
    const __bf16* __restrict__ B, int N, int K,
    const float* __restrict__ bias, const float* __restrict__ res,
    float* __restrict__ out) {
  int NT = N >> 6;
  int gw = (blockIdx.x * blockDim.x + threadIdx.x) >> 5;
  int lane = threadIdx.x & 31;
  int mt = gw / NT, nt = gw - mt * NT;
  if (mt >= TOKENS / 32) return;
  v8f z = {};
  v8f acc[8] = {z, z, z, z, z, z, z, z};
  gemm_tile_32x64(A + (size_t)(mt * 32) * lda, lda, B, N >> 4, nt * 4, K, lane, acc);
  int n = lane & 15, h = lane >> 4;
#pragma unroll
  for (int rt = 0; rt < 2; ++rt) {
#pragma unroll
    for (int vv = 0; vv < 8; ++vv) {
      int m = mt * 32 + rt * 16 + vv + 8 * h;
#pragma unroll
      for (int j = 0; j < 4; ++j) {
        int col = nt * 64 + j * 16 + n;
        size_t idx = (size_t)m * N + col;
        out[idx] = res[idx] + acc[rt * 4 + j][vv] + bias[col];
      }
    }
  }
}

// out[bf16, M x N] = relu(A.Bswz + bias)   (MLP1)
__global__ void __launch_bounds__(256) gemm_relu_kernel(
    const __bf16* __restrict__ A, int lda,
    const __bf16* __restrict__ B, int N, int K,
    const float* __restrict__ bias, __bf16* __restrict__ out) {
  int NT = N >> 6;
  int gw = (blockIdx.x * blockDim.x + threadIdx.x) >> 5;
  int lane = threadIdx.x & 31;
  int mt = gw / NT, nt = gw - mt * NT;
  if (mt >= TOKENS / 32) return;
  v8f z = {};
  v8f acc[8] = {z, z, z, z, z, z, z, z};
  gemm_tile_32x64(A + (size_t)(mt * 32) * lda, lda, B, N >> 4, nt * 4, K, lane, acc);
  int n = lane & 15, h = lane >> 4;
#pragma unroll
  for (int rt = 0; rt < 2; ++rt) {
#pragma unroll
    for (int vv = 0; vv < 8; ++vv) {
      int m = mt * 32 + rt * 16 + vv + 8 * h;
#pragma unroll
      for (int j = 0; j < 4; ++j) {
        int col = nt * 64 + j * 16 + n;
        out[(size_t)m * N + col] = f2bf(fmaxf(acc[rt * 4 + j][vv] + bias[col], 0.0f));
      }
    }
  }
}

// ---------------------------------------------------------------- launch

extern "C" void kernel_launch(void* const* d_in, const int* in_sizes, int n_in,
                              void* d_out, int out_size, void* d_ws, size_t ws_size,
                              hipStream_t stream) {
  (void)in_sizes; (void)n_in; (void)out_size; (void)ws_size;
  const float* x   = (const float*)d_in[0];
  const float* Wq  = (const float*)d_in[1];
  const float* bq  = (const float*)d_in[2];
  const float* Wk  = (const float*)d_in[3];
  const float* bk  = (const float*)d_in[4];
  const float* Wv  = (const float*)d_in[5];
  const float* bv  = (const float*)d_in[6];
  const float* Wp  = (const float*)d_in[7];
  const float* bp  = (const float*)d_in[8];
  const float* W1  = (const float*)d_in[9];
  const float* b1  = (const float*)d_in[10];
  const float* W2  = (const float*)d_in[11];
  const float* b2  = (const float*)d_in[12];
  const float* g1  = (const float*)d_in[13];
  const float* be1 = (const float*)d_in[14];
  const float* g2  = (const float*)d_in[15];
  const float* be2 = (const float*)d_in[16];
  float* out = (float*)d_out;

  char* ws = (char*)d_ws;
  __bf16* wqkv = (__bf16*)(ws + 0);          // 512x1536 swz (1.5 MB)
  __bf16* wp   = (__bf16*)(ws + 1572864);    // 512x512 swz  (0.5 MB)
  __bf16* w1   = (__bf16*)(ws + 2097152);    // 512x2048 swz (2 MB)
  __bf16* w2   = (__bf16*)(ws + 4194304);    // 2048x512 swz (2 MB)
  __bf16* xn   = (__bf16*)(ws + 6291456);    // 4096x512 row-major (4 MB, reused LN2)
  float*  x1   = (float*) (ws + 10485760);   // 4096x512 f32 (8 MB)
  __bf16* qb   = (__bf16*)(ws + 18874368);   // [B,H] A-swz  (4 MB)
  __bf16* kTb  = (__bf16*)(ws + 23068672);   // [B,H] B-swz  (4 MB)
  __bf16* vb   = (__bf16*)(ws + 27262976);   // [B,H] B-swz  (4 MB)
  __bf16* ob   = (__bf16*)(ws + 31457280);   // 4096x512 row-major (4 MB)
  __bf16* hb   = (__bf16*)(ws + 18874368);   // 4096x2048 (16 MB, overlays q/kT/v/o)

  // weights -> bf16, fragment-swizzled
  cvt_qkv_kernel<<<256, 256, 0, stream>>>(Wq, Wk, Wv, wqkv);
  cvt_swz_kernel<<<256, 256, 0, stream>>>(Wp, wp, DMODEL, DMODEL);
  cvt_swz_kernel<<<256, 256, 0, stream>>>(W1, w1, DMODEL, DFF);
  cvt_swz_kernel<<<256, 256, 0, stream>>>(W2, w2, DFF, DMODEL);

  // attention sublayer
  ln_kernel<<<TOKENS, 256, 0, stream>>>(x, g1, be1, xn);
  gemm_qkv_kernel<<<384, 256, 0, stream>>>(xn, wqkv, bq, bk, bv, qb, kTb, vb);
  attn_kernel<<<512, 128, 0, stream>>>(qb, kTb, vb, ob);
  gemm_bias_res_kernel<<<128, 256, 0, stream>>>(ob, DMODEL, wp, DMODEL, DMODEL,
                                                bp, x, x1);
  // MLP sublayer
  ln_kernel<<<TOKENS, 256, 0, stream>>>(x1, g2, be2, xn);
  gemm_relu_kernel<<<512, 256, 0, stream>>>(xn, DMODEL, w1, DFF, DMODEL, b1, hb);
  gemm_bias_res_kernel<<<128, 256, 0, stream>>>(hb, DFF, w2, DMODEL, DFF,
                                                b2, x1, out);
}
